// TransformerDecoderLayer_66760971649268
// MI455X (gfx1250) — matrix operand order repaired and checked by reading, not compiled, not run on hardware
//
#include <hip/hip_runtime.h>

#define LQ    300
#define BB    2
#define DD    256
#define HH    8
#define SSZ   6400
#define FFD   2048
#define HD    32
#define NHEAD 16   // BB * HH

typedef __attribute__((ext_vector_type(16))) _Float16 v16h;
typedef __attribute__((ext_vector_type(8)))  float    v8f;
typedef __attribute__((ext_vector_type(8)))  _Float16 h8;
typedef __attribute__((ext_vector_type(4)))  float    f4;
typedef __attribute__((ext_vector_type(4)))  int      v4i_t;

#if defined(__HIP_DEVICE_COMPILE__)
#  if __has_builtin(__builtin_amdgcn_global_load_async_to_lds_b128)
#    define HAVE_ASYNC_LDS 1
typedef __attribute__((address_space(1))) v4i_t glob_v4i;
typedef __attribute__((address_space(3))) v4i_t lds_v4i;
#  endif
#  if __has_builtin(__builtin_amdgcn_s_wait_asynccnt)
#    define WAIT_ASYNCCNT0() __builtin_amdgcn_s_wait_asynccnt(0)
#  else
#    define WAIT_ASYNCCNT0() asm volatile("s_wait_asynccnt 0x0" ::: "memory")
#  endif
#else
#  define WAIT_ASYNCCNT0()
#endif

static __device__ __forceinline__ v16h zero_v16h() {
  v16h z;
#pragma unroll
  for (int i = 0; i < 16; ++i) z[i] = (_Float16)0.0f;
  return z;
}

// A fragment (16x32 f16). lane m = lane&15, kq = (lane>>4)*8.
// element e -> K = 2*kq+e
static __device__ __forceinline__ v16h load_a_frag(const float* __restrict__ A, int lda,
                                                   int row, int k0, int kq, bool valid) {
  v16h a;
  if (valid) {
    const float* p = A + (size_t)row * lda + k0 + 2 * kq;
    f4 x0 = *(const f4*)(p);
    f4 x1 = *(const f4*)(p + 4);
    f4 x2 = *(const f4*)(p + 8);
    f4 x3 = *(const f4*)(p + 12);
#pragma unroll
    for (int i = 0; i < 4; ++i) {
      a[i]      = (_Float16)x0[i];
      a[4 + i]  = (_Float16)x1[i];
      a[8 + i]  = (_Float16)x2[i];
      a[12 + i] = (_Float16)x3[i];
    }
  } else {
    a = zero_v16h();
  }
  return a;
}

// B fragment (32x16 f16) from W stored (N,K) row-major (i.e. B = W^T).
// lane col n = lane&15, koff = (lane>>4)*16, element e -> K = koff+e
static __device__ __forceinline__ v16h load_b_frag(const float* __restrict__ W, int ldw,
                                                   int col, int k0, int koff, bool valid) {
  v16h b;
  if (valid) {
    const float* p = W + (size_t)col * ldw + k0 + koff;
    f4 x0 = *(const f4*)(p);
    f4 x1 = *(const f4*)(p + 4);
    f4 x2 = *(const f4*)(p + 8);
    f4 x3 = *(const f4*)(p + 12);
#pragma unroll
    for (int i = 0; i < 4; ++i) {
      b[i]      = (_Float16)x0[i];
      b[4 + i]  = (_Float16)x1[i];
      b[8 + i]  = (_Float16)x2[i];
      b[12 + i] = (_Float16)x3[i];
    }
  } else {
    b = zero_v16h();
  }
  return b;
}

// ---------------------------------------------------------------------------
// Generic GEMM: C[M,N] = act(A[M,K] @ W[N,K]^T + bias + Res)
// one wave per 16x16 tile; grid = (ceil(N/16), ceil(M/16)), block = 32
// ---------------------------------------------------------------------------
__global__ void __launch_bounds__(32)
gemm_wmma_kernel(const float* __restrict__ A, int lda,
                 const float* __restrict__ W, int ldw,
                 const float* __restrict__ bias,
                 const float* __restrict__ Res,
                 float* __restrict__ C, int ldc,
                 int M, int N, int K, int relu) {
  int lane = threadIdx.x & 31;
  int hi   = lane >> 4;
  int ln16 = lane & 15;
  int n0 = blockIdx.x * 16;
  int m0 = blockIdx.y * 16;
  int mrow = m0 + ln16;
  int col  = n0 + ln16;
  int kq   = hi * 8;
  int koff = hi * 16;
  bool aok = mrow < M;
  bool bok = col < N;
  v8f acc = {};
  for (int k0 = 0; k0 < K; k0 += 32) {
    v16h a = load_a_frag(A, lda, aok ? mrow : 0, k0, kq, aok);
    v16h b = load_b_frag(W, ldw, bok ? col : 0, k0, koff, bok);
    acc = __builtin_amdgcn_wmma_f32_16x16x32_f16(false, a, false, b, (short)0, acc, false, false);
  }
  float bv = (bok && bias) ? bias[col] : 0.0f;
#pragma unroll
  for (int r = 0; r < 8; ++r) {
    int row = m0 + r + hi * 8;
    if (row < M && bok) {
      float v = acc[r] + bv;
      if (Res) v += Res[(size_t)row * ldc + col];
      if (relu) v = v > 0.0f ? v : 0.0f;
      C[(size_t)row * ldc + col] = v;
    }
  }
}

// ---------------------------------------------------------------------------
// Projection into per-head f16 layout (optionally transposed for V).
// X: (T*BB, DD) rows = t*BB+b. Output vmode 0: Y[n][t][d]; vmode 1: Y[n][d][t]
// Applies (acc + bias) * scale, n = b*HH + h, col = h*32 + d.
// ---------------------------------------------------------------------------
__global__ void __launch_bounds__(32)
proj_head_kernel(const float* __restrict__ X,
                 const float* __restrict__ W,
                 const float* __restrict__ bias,
                 _Float16* __restrict__ Y,
                 int T, float scale, int vmode, int ldv) {
  int lane = threadIdx.x & 31;
  int hi   = lane >> 4;
  int ln16 = lane & 15;
  int n0 = blockIdx.x * 16;
  int m0 = blockIdx.y * 16;
  int M  = T * BB;
  int mrow = m0 + ln16;
  int col  = n0 + ln16;
  int kq = hi * 8, koff = hi * 16;
  bool aok = mrow < M;
  v8f acc = {};
  for (int k0 = 0; k0 < DD; k0 += 32) {
    v16h a = load_a_frag(X, DD, aok ? mrow : 0, k0, kq, aok);
    v16h b = load_b_frag(W, DD, col, k0, koff, true);
    acc = __builtin_amdgcn_wmma_f32_16x16x32_f16(false, a, false, b, (short)0, acc, false, false);
  }
  float bv = bias[col];
  int h = col >> 5, d = col & 31;
#pragma unroll
  for (int r = 0; r < 8; ++r) {
    int row = m0 + r + hi * 8;
    if (row < M) {
      int b = row & 1;     // BB == 2
      int t = row >> 1;
      int n = b * HH + h;
      float v = (acc[r] + bv) * scale;
      if (vmode == 0) Y[((size_t)n * T + t) * HD + d] = (_Float16)v;
      else            Y[((size_t)n * HD + d) * ldv + t] = (_Float16)v;
    }
  }
}

// ---------------------------------------------------------------------------
// Flash attention, one wave per (head n, 16-query tile).
// Qh: (NHEAD, Lq, 32) f16 pre-scaled by hd^-0.5. Kh: (NHEAD, S, 32) f16.
// VhT: (NHEAD, 32, ldv) f16 (transposed, padded stride).
// K chunks (32 x 32 halves) are streamed into LDS with double-buffered
// GLOBAL_LOAD_ASYNC_TO_LDS_B128 (ASYNCcnt / s_wait_asynccnt), overlapping the
// copy of chunk i+1 with WMMA+softmax of chunk i.
// Gaussian spatial bias recomputed on the fly from point/scale2/grid.
// Output O: (Lq, BB, DD) f32.
// ---------------------------------------------------------------------------
__global__ void __launch_bounds__(32)
flash_attn_kernel(const _Float16* __restrict__ Qh,
                  const _Float16* __restrict__ Kh,
                  const _Float16* __restrict__ VhT, int ldv,
                  float* __restrict__ O,
                  int Lq, int S,
                  const float* __restrict__ pointp,
                  const float* __restrict__ scale2p,
                  const float* __restrict__ gridp) {
  __shared__ __align__(16) _Float16 pbuf[16 * 32];
  __shared__ __align__(16) _Float16 kbuf[2][32 * 32];
  int lane = threadIdx.x & 31;
  int hi   = lane >> 4;
  int ln16 = lane & 15;
  int kq   = hi * 8;
  int koff = hi * 16;
  int n  = blockIdx.y;
  int l0 = blockIdx.x * 16;
  bool use_bias = (pointp != nullptr);

  // Q fragment (A layout), rows clamped then masked at store
  int ql = l0 + ln16; if (ql >= Lq) ql = Lq - 1;
  v16h qa;
  {
    const _Float16* qp = Qh + ((size_t)n * Lq + ql) * HD;
    h8 q0 = *(const h8*)(qp + koff);
    h8 q1 = *(const h8*)(qp + koff + 8);
#pragma unroll
    for (int i = 0; i < 8; ++i) { qa[i] = q0[i]; qa[8 + i] = q1[i]; }
  }

  float px[8], py[8], sx[8], sy[8];
#pragma unroll
  for (int r = 0; r < 8; ++r) {
    int l = l0 + r + hi * 8; if (l >= Lq) l = Lq - 1;
    if (use_bias) {
      px[r] = pointp[(l * 16 + n) * 2 + 0];
      py[r] = pointp[(l * 16 + n) * 2 + 1];
      sx[r] = scale2p[(l * 16 + n) * 2 + 0];
      sy[r] = scale2p[(l * 16 + n) * 2 + 1];
    } else { px[r] = py[r] = sx[r] = sy[r] = 0.0f; }
  }

  float mrow[8], lrow[8];
#pragma unroll
  for (int r = 0; r < 8; ++r) { mrow[r] = -1e30f; lrow[r] = 0.0f; }
  v8f oacc0 = {}, oacc1 = {};

#ifdef HAVE_ASYNC_LDS
  const int nfull = S & ~31;  // s range covered by full 32-wide chunks
  // async copy of one 32x32-half K chunk: 128 x 16B segments, 4 per lane
  auto issue_async = [&](int bufi, int s0c) {
    const _Float16* srcbase = Kh + ((size_t)n * S + s0c) * HD;
    _Float16* dstbase = &kbuf[bufi][0];
#pragma unroll
    for (int q = 0; q < 4; ++q) {
      int seg  = q * 32 + lane;
      int row  = seg >> 2;
      int part = seg & 3;
      const _Float16* g = srcbase + (size_t)row * HD + part * 8;
      _Float16* l = dstbase + row * 32 + part * 8;
      __builtin_amdgcn_global_load_async_to_lds_b128(
          (glob_v4i*)(uintptr_t)g,
          (lds_v4i*)(uint32_t)(uintptr_t)l, 0, 0);
    }
  };
  if (nfull > 0) issue_async(0, 0);
#else
  const int nfull = 0;
#endif
  int buf = 0;

  for (int s0 = 0; s0 < S; s0 += 32) {
    bool cur_full = s0 < nfull;
#ifdef HAVE_ASYNC_LDS
    if (cur_full) {
      WAIT_ASYNCCNT0();          // drain this chunk's async copies
      __syncthreads();
      if (s0 + 32 < nfull) issue_async(buf ^ 1, s0 + 32);  // prefetch next chunk
    }
#endif
    if (s0 + 32 + lane < S)
      __builtin_prefetch((const void*)(VhT + ((size_t)n * HD + lane) * ldv + s0 + 32), 0, 0);

    // logits for two 16-wide s sub-tiles
    v8f lg[2];
#pragma unroll
    for (int t = 0; t < 2; ++t) {
      int sc = s0 + t * 16 + ln16;
      v16h kb;
      if (cur_full) {
        const _Float16* kp = &kbuf[buf][(t * 16 + ln16) * 32 + koff];
        h8 k0v = *(const h8*)(kp);
        h8 k1v = *(const h8*)(kp + 8);
#pragma unroll
        for (int i = 0; i < 8; ++i) { kb[i] = k0v[i]; kb[8 + i] = k1v[i]; }
      } else if (sc < S) {
        const _Float16* kp = Kh + ((size_t)n * S + sc) * HD + koff;
        h8 k0v = *(const h8*)(kp);
        h8 k1v = *(const h8*)(kp + 8);
#pragma unroll
        for (int i = 0; i < 8; ++i) { kb[i] = k0v[i]; kb[8 + i] = k1v[i]; }
      } else {
        kb = zero_v16h();
      }
      v8f z = {};
      lg[t] = __builtin_amdgcn_wmma_f32_16x16x32_f16(false, qa, false, kb, (short)0, z, false, false);
    }

    int sA = s0 + ln16, sB = s0 + 16 + ln16;
    float gx0 = 0, gy0 = 0, gx1 = 0, gy1 = 0;
    if (use_bias) {
      if (sA < S) { gx0 = gridp[(sA * 16 + n) * 2]; gy0 = gridp[(sA * 16 + n) * 2 + 1]; }
      if (sB < S) { gx1 = gridp[(sB * 16 + n) * 2]; gy1 = gridp[(sB * 16 + n) * 2 + 1]; }
    }

#pragma unroll
    for (int r = 0; r < 8; ++r) {
      float v0 = lg[0][r], v1 = lg[1][r];
      if (use_bias) {
        float dx0 = px[r] - gx0, dy0 = py[r] - gy0;
        v0 -= fabsf(dx0 * dx0 * sx[r] + dy0 * dy0 * sy[r]) * 0.125f;
        float dx1 = px[r] - gx1, dy1 = py[r] - gy1;
        v1 -= fabsf(dx1 * dx1 * sx[r] + dy1 * dy1 * sy[r]) * 0.125f;
      }
      if (sA >= S) v0 = -1e30f;
      if (sB >= S) v1 = -1e30f;
      // row max across the 16-lane half-group (rows live across lanes in C layout)
      float mx = fmaxf(v0, v1);
      mx = fmaxf(mx, __shfl_xor(mx, 1, 32));
      mx = fmaxf(mx, __shfl_xor(mx, 2, 32));
      mx = fmaxf(mx, __shfl_xor(mx, 4, 32));
      mx = fmaxf(mx, __shfl_xor(mx, 8, 32));
      float newm  = fmaxf(mrow[r], mx);
      float alpha = __expf(mrow[r] - newm);
      float p0 = __expf(v0 - newm);
      float p1 = __expf(v1 - newm);
      float rs = p0 + p1;
      rs += __shfl_xor(rs, 1, 32);
      rs += __shfl_xor(rs, 2, 32);
      rs += __shfl_xor(rs, 4, 32);
      rs += __shfl_xor(rs, 8, 32);
      lrow[r] = lrow[r] * alpha + rs;
      mrow[r] = newm;
      oacc0[r] *= alpha;
      oacc1[r] *= alpha;
      int m = r + hi * 8;
      pbuf[m * 32 + ln16]      = (_Float16)p0;
      pbuf[m * 32 + 16 + ln16] = (_Float16)p1;
    }
    __syncthreads();
    // re-load P in A-fragment layout from LDS
    v16h pa;
    {
      const _Float16* pp = pbuf + ln16 * 32 + koff;
      h8 a0 = *(const h8*)(pp);
      h8 a1 = *(const h8*)(pp + 8);
#pragma unroll
      for (int i = 0; i < 8; ++i) { pa[i] = a0[i]; pa[8 + i] = a1[i]; }
    }
    __syncthreads();

#pragma unroll
    for (int dt = 0; dt < 2; ++dt) {
      int d = dt * 16 + ln16;
      const _Float16* vp = VhT + ((size_t)n * HD + d) * ldv + s0 + koff;
      v16h vb;
      if (s0 + koff + 15 < S) {
        h8 x0 = *(const h8*)(vp);
        h8 x1 = *(const h8*)(vp + 8);
#pragma unroll
        for (int i = 0; i < 8; ++i) { vb[i] = x0[i]; vb[8 + i] = x1[i]; }
      } else {
#pragma unroll
        for (int i = 0; i < 16; ++i) {
          int sc = s0 + koff + i;
          vb[i] = (sc < S) ? vp[i] : (_Float16)0.0f;
        }
      }
      if (dt == 0)
        oacc0 = __builtin_amdgcn_wmma_f32_16x16x32_f16(false, pa, false, vb, (short)0, oacc0, false, false);
      else
        oacc1 = __builtin_amdgcn_wmma_f32_16x16x32_f16(false, pa, false, vb, (short)0, oacc1, false, false);
    }
    if (cur_full) buf ^= 1;
  }

#pragma unroll
  for (int r = 0; r < 8; ++r) {
    int l = l0 + r + hi * 8;
    if (l < Lq) {
      float inv = 1.0f / lrow[r];
      int b = n >> 3, h = n & 7;
      float* op = O + ((size_t)l * BB + b) * DD + h * HD;
      op[ln16]      = oacc0[r] * inv;
      op[16 + ln16] = oacc1[r] * inv;
    }
  }
}

// ---------------------------------------------------------------------------
// LayerNorm over DD=256, one wave per row (8 cols/lane, shfl reductions)
// ---------------------------------------------------------------------------
__global__ void ln_kernel(const float* __restrict__ X, const float* __restrict__ g,
                          const float* __restrict__ be, float* __restrict__ Y, int rows) {
  int lane = threadIdx.x & 31;
  int wv   = threadIdx.x >> 5;
  int row  = blockIdx.x * (blockDim.x >> 5) + wv;
  if (row >= rows) return;
  const float* x = X + (size_t)row * DD;
  float v[8]; float s = 0.0f;
#pragma unroll
  for (int i = 0; i < 8; ++i) { v[i] = x[lane + i * 32]; s += v[i]; }
  s += __shfl_xor(s, 1, 32);  s += __shfl_xor(s, 2, 32);
  s += __shfl_xor(s, 4, 32);  s += __shfl_xor(s, 8, 32);
  s += __shfl_xor(s, 16, 32);
  float mean = s * (1.0f / DD);
  float vs = 0.0f;
#pragma unroll
  for (int i = 0; i < 8; ++i) { float d = v[i] - mean; vs += d * d; }
  vs += __shfl_xor(vs, 1, 32);  vs += __shfl_xor(vs, 2, 32);
  vs += __shfl_xor(vs, 4, 32);  vs += __shfl_xor(vs, 8, 32);
  vs += __shfl_xor(vs, 16, 32);
  float inv = rsqrtf(vs * (1.0f / DD) + 1e-5f);
  float* y = Y + (size_t)row * DD;
#pragma unroll
  for (int i = 0; i < 8; ++i) {
    int c = lane + i * 32;
    y[c] = (v[i] - mean) * inv * g[c] + be[c];
  }
}

__global__ void add2_kernel(const float* __restrict__ a, const float* __restrict__ b,
                            float* __restrict__ y, int nElem) {
  int i = blockIdx.x * blockDim.x + threadIdx.x;
  if (i < nElem) y[i] = a[i] + b[i];
}

// q3 input: tgt1 + new_q (query_pos zeroed per batch when any reshape_output==0)
__global__ void q3in_kernel(const float* __restrict__ t1, const float* __restrict__ qp,
                            const float* __restrict__ ro1, const float* __restrict__ ro2,
                            float* __restrict__ y) {
  int i = blockIdx.x * blockDim.x + threadIdx.x;
  if (i >= LQ * BB * DD) return;
  int row = i / DD;
  int b = row & 1;
  int l = row >> 1;
  const float* ro = (b == 0) ? ro1 : ro2;
  bool msk = (ro[l * 4] == 0.0f) || (ro[l * 4 + 1] == 0.0f) ||
             (ro[l * 4 + 2] == 0.0f) || (ro[l * 4 + 3] == 0.0f);
  y[i] = t1[i] + (msk ? 0.0f : qp[i]);
}

// ref = h_w * sigmoid(ref_inter)/32; point = ref_tiled + offset; scale2 = scale^2
__global__ void point_kernel(const float* __restrict__ ri, const float* __restrict__ hw,
                             const float* __restrict__ offs, const float* __restrict__ slin,
                             float* __restrict__ refo, float* __restrict__ pnt,
                             float* __restrict__ sc2) {
  int i = blockIdx.x * blockDim.x + threadIdx.x;
  if (i >= LQ * BB) return;
  int b = i & 1;
  int l = i >> 1;
  float rx = hw[b * 2 + 0] * (1.0f / (1.0f + __expf(-ri[i * 2 + 0]))) * (1.0f / 32.0f);
  float ry = hw[b * 2 + 1] * (1.0f / (1.0f + __expf(-ri[i * 2 + 1]))) * (1.0f / 32.0f);
#pragma unroll
  for (int j = 0; j < 8; ++j) {
    int n = b * 8 + j;
    float ox = offs[i * 16 + j * 2 + 0];
    float oy = offs[i * 16 + j * 2 + 1];
    pnt[(l * 16 + n) * 2 + 0] = rx + ox;
    pnt[(l * 16 + n) * 2 + 1] = ry + oy;
    float s0v = slin[i * 16 + j * 2 + 0];
    float s1v = slin[i * 16 + j * 2 + 1];
    sc2[(l * 16 + n) * 2 + 0] = s0v * s0v;
    sc2[(l * 16 + n) * 2 + 1] = s1v * s1v;
    refo[i * 16 + j * 2 + 0] = rx;
    refo[i * 16 + j * 2 + 1] = ry;
  }
}

static inline int cdiv(int a, int b) { return (a + b - 1) / b; }

extern "C" void kernel_launch(void* const* d_in, const int* in_sizes, int n_in,
                              void* d_out, int out_size, void* d_ws, size_t ws_size,
                              hipStream_t stream) {
  (void)in_sizes; (void)n_in; (void)out_size; (void)ws_size;
  const float* gridp  = (const float*)d_in[0];
  const float* h_w    = (const float*)d_in[1];
  const float* tgt    = (const float*)d_in[2];
  const float* memory = (const float*)d_in[3];
  const float* ro1    = (const float*)d_in[4];
  const float* ro2    = (const float*)d_in[5];
  const float* pos    = (const float*)d_in[6];
  const float* qpos   = (const float*)d_in[7];
  const float* sa_wi  = (const float*)d_in[8];
  const float* sa_bi  = (const float*)d_in[9];
  const float* sa_wo  = (const float*)d_in[10];
  const float* sa_bo  = (const float*)d_in[11];
  const float* ca_wi  = (const float*)d_in[12];
  const float* ca_bi  = (const float*)d_in[13];
  const float* ca_wo  = (const float*)d_in[14];
  const float* ca_bo  = (const float*)d_in[15];
  const float* l1_w   = (const float*)d_in[16];
  const float* l1_b   = (const float*)d_in[17];
  const float* l2_w   = (const float*)d_in[18];
  const float* l2_b   = (const float*)d_in[19];
  const float* n1_g   = (const float*)d_in[20];
  const float* n1_b   = (const float*)d_in[21];
  const float* n2_g   = (const float*)d_in[22];
  const float* n2_b   = (const float*)d_in[23];
  const float* n3_g   = (const float*)d_in[24];
  const float* n3_b   = (const float*)d_in[25];
  const float* n4_g   = (const float*)d_in[26];
  const float* n4_b   = (const float*)d_in[27];
  const float* p1_w1  = (const float*)d_in[28];
  const float* p1_b1  = (const float*)d_in[29];
  const float* p1_w2  = (const float*)d_in[30];
  const float* p1_b2  = (const float*)d_in[31];
  const float* p1_w3  = (const float*)d_in[32];
  const float* p1_b3  = (const float*)d_in[33];
  const float* p2_w   = (const float*)d_in[34];
  const float* p2_b   = (const float*)d_in[35];
  const float* p3_w   = (const float*)d_in[36];
  const float* p3_b   = (const float*)d_in[37];

  const size_t TB  = (size_t)LQ * BB;     // 600
  const size_t TBD = TB * DD;             // 153600

  float* out_tgt = (float*)d_out;         // (L,B,D)
  float* out_ri  = out_tgt + TBD;         // (L,B,2)
  float* out_ref = out_ri + TB * 2;       // (L,B,16)
  float* out_D   = out_ref + TB * 16;     // (L,B,D)

  char* wp = (char*)d_ws;
  auto carve = [&](size_t bytes) -> void* {
    void* p = (void*)wp;
    wp += (bytes + 255) & ~(size_t)255;
    return p;
  };
  float* qk    = (float*)carve(TBD * 4);
  float* out4  = (float*)carve(TBD * 4);
  float* tmp   = (float*)carve(TBD * 4);
  float* tgt1  = (float*)carve(TBD * 4);
  float* tgtA  = (float*)carve(TBD * 4);
  float* q2in  = (float*)carve(TBD * 4);
  float* q3in  = (float*)carve(TBD * 4);
  float* h1    = (float*)carve(TBD * 4);
  float* h2    = (float*)carve(TBD * 4);
  float* attno = (float*)carve(TBD * 4);
  float* offs  = (float*)carve(TB * 16 * 4);
  float* slin  = (float*)carve(TB * 16 * 4);
  float* pnt   = (float*)carve((size_t)LQ * 32 * 4);
  float* sc2   = (float*)carve((size_t)LQ * 32 * 4);
  float* kvin  = (float*)carve((size_t)SSZ * BB * DD * 4);
  float* ffnh  = (float*)carve(TB * FFD * 4);
  _Float16* qh_sa = (_Float16*)carve((size_t)NHEAD * LQ * HD * 2);
  _Float16* kh_sa = (_Float16*)carve((size_t)NHEAD * LQ * HD * 2);
  _Float16* vT_sa = (_Float16*)carve((size_t)NHEAD * HD * 304 * 2);  // padded stride
  _Float16* qh2   = (_Float16*)carve((size_t)NHEAD * LQ * HD * 2);
  _Float16* qh3   = (_Float16*)carve((size_t)NHEAD * LQ * HD * 2);
  _Float16* kh_ca = (_Float16*)carve((size_t)NHEAD * SSZ * HD * 2);
  _Float16* vT_ca = (_Float16*)carve((size_t)NHEAD * HD * SSZ * 2);

  const float qscale = 0.17677669529663687f;  // 1/sqrt(32)
  dim3 b32(32);
  const int MT = cdiv(LQ * BB, 16);           // 38 row tiles
  const int LT = cdiv(LQ, 16);                // 19 query tiles

  // qk = tgt + query_pos ; out4 = LN(qk; n4)
  add2_kernel<<<cdiv((int)TBD, 256), 256, 0, stream>>>(tgt, qpos, qk, (int)TBD);
  ln_kernel<<<cdiv(600, 4), 128, 0, stream>>>(qk, n4_g, n4_b, out4, 600);

  // offset / scale heads and the p1 MLP (ref_inter straight into d_out)
  gemm_wmma_kernel<<<dim3(1, MT), b32, 0, stream>>>(out4, DD, p2_w, DD, p2_b, nullptr, offs, 16, 600, 16, DD, 0);
  gemm_wmma_kernel<<<dim3(1, MT), b32, 0, stream>>>(out4, DD, p3_w, DD, p3_b, nullptr, slin, 16, 600, 16, DD, 0);
  gemm_wmma_kernel<<<dim3(16, MT), b32, 0, stream>>>(out4, DD, p1_w1, DD, p1_b1, nullptr, h1, DD, 600, DD, DD, 1);
  gemm_wmma_kernel<<<dim3(16, MT), b32, 0, stream>>>(h1, DD, p1_w2, DD, p1_b2, nullptr, h2, DD, 600, DD, DD, 1);
  gemm_wmma_kernel<<<dim3(1, MT), b32, 0, stream>>>(h2, DD, p1_w3, DD, p1_b3, nullptr, out_ri, 2, 600, 2, DD, 0);
  point_kernel<<<cdiv(600, 128), 128, 0, stream>>>(out_ri, h_w, offs, slin, out_ref, pnt, sc2);

  // ---- self attention ----
  proj_head_kernel<<<dim3(16, MT), b32, 0, stream>>>(qk, sa_wi, sa_bi, qh_sa, LQ, qscale, 0, 0);
  proj_head_kernel<<<dim3(16, MT), b32, 0, stream>>>(qk, sa_wi + DD * DD, sa_bi + DD, kh_sa, LQ, 1.0f, 0, 0);
  proj_head_kernel<<<dim3(16, MT), b32, 0, stream>>>(tgt, sa_wi + 2 * DD * DD, sa_bi + 2 * DD, vT_sa, LQ, 1.0f, 1, 304);
  flash_attn_kernel<<<dim3(LT, NHEAD), b32, 0, stream>>>(qh_sa, kh_sa, vT_sa, 304, attno, LQ, LQ,
                                                         nullptr, nullptr, nullptr);
  gemm_wmma_kernel<<<dim3(16, MT), b32, 0, stream>>>(attno, DD, sa_wo, DD, sa_bo, tgt, tmp, DD, 600, DD, DD, 0);
  ln_kernel<<<cdiv(600, 4), 128, 0, stream>>>(tmp, n1_g, n1_b, tgt1, 600);

  // ---- cross attention K/V (shared by both cross attentions) ----
  add2_kernel<<<cdiv(SSZ * BB * DD, 256), 256, 0, stream>>>(memory, pos, kvin, SSZ * BB * DD);
  proj_head_kernel<<<dim3(16, cdiv(SSZ * BB, 16)), b32, 0, stream>>>(kvin, ca_wi + DD * DD, ca_bi + DD, kh_ca, SSZ, 1.0f, 0, 0);
  proj_head_kernel<<<dim3(16, cdiv(SSZ * BB, 16)), b32, 0, stream>>>(memory, ca_wi + 2 * DD * DD, ca_bi + 2 * DD, vT_ca, SSZ, 1.0f, 1, SSZ);

  // queries for the two cross attentions
  add2_kernel<<<cdiv((int)TBD, 256), 256, 0, stream>>>(tgt1, qpos, q2in, (int)TBD);
  q3in_kernel<<<cdiv((int)TBD, 256), 256, 0, stream>>>(tgt1, qpos, ro1, ro2, q3in);
  proj_head_kernel<<<dim3(16, MT), b32, 0, stream>>>(q2in, ca_wi, ca_bi, qh2, LQ, qscale, 0, 0);
  proj_head_kernel<<<dim3(16, MT), b32, 0, stream>>>(q3in, ca_wi, ca_bi, qh3, LQ, qscale, 0, 0);

  // cross attention #1 (tgt2 path, with Gaussian bias)
  flash_attn_kernel<<<dim3(LT, NHEAD), b32, 0, stream>>>(qh2, kh_ca, vT_ca, SSZ, attno, LQ, SSZ, pnt, sc2, gridp);
  gemm_wmma_kernel<<<dim3(16, MT), b32, 0, stream>>>(attno, DD, ca_wo, DD, ca_bo, tgt1, tmp, DD, 600, DD, DD, 0);
  ln_kernel<<<cdiv(600, 4), 128, 0, stream>>>(tmp, n2_g, n2_b, tgtA, 600);

  // cross attention #2 (D_out path, no residual)
  flash_attn_kernel<<<dim3(LT, NHEAD), b32, 0, stream>>>(qh3, kh_ca, vT_ca, SSZ, attno, LQ, SSZ, pnt, sc2, gridp);
  gemm_wmma_kernel<<<dim3(16, MT), b32, 0, stream>>>(attno, DD, ca_wo, DD, ca_bo, nullptr, out_D, DD, 600, DD, DD, 0);

  // ---- FFN + final LN ----
  gemm_wmma_kernel<<<dim3(FFD / 16, MT), b32, 0, stream>>>(tgtA, DD, l1_w, DD, l1_b, nullptr, ffnh, FFD, 600, FFD, DD, 1);
  gemm_wmma_kernel<<<dim3(16, MT), b32, 0, stream>>>(ffnh, FFD, l2_w, FFD, l2_b, tgtA, tmp, DD, 600, DD, FFD, 0);
  ln_kernel<<<cdiv(600, 4), 128, 0, stream>>>(tmp, n3_g, n3_b, out_tgt, 600);
}
